// GraphConvLayer_19078244729007
// MI455X (gfx1250) — matrix operand order repaired and checked by reading, not compiled, not run on hardware
//
#include <hip/hip_runtime.h>
#include <hip/hip_bf16.h>

// ---------------------------------------------------------------------------
// GraphConv layer for MI455X (gfx1250, wave32, WMMA).
//
// Fusion:  h only feeds msg layer1  =>  fold node_W@W1h and edge_W@W1e.
//   hm   = node_feats @ (node_W @ msg_W1[:H])  + all biases   [N,H]   (ws)
//   pre  = hm[src] + edge_feats @ (edge_W @ msg_W1[H:])       [E,H]
//   m    = relu(pre) @ msg_W2 + msg_b2                         [E,H]
//   agg  = scatter-add(m, dst)   (native f32 global atomics, L2-resident)
//   out  = relu([nf,agg] @ upd_W1 + b1) @ upd_W2 + b2
//
// All GEMMs: v_wmma_f32_16x16x32_f16 (f16 A/B, f32 accum). One 16-row tile
// per wave, 8 waves per block.
//
// NEW vs round 1: all B matrices are pre-transposed + pre-converted into
// their exact f16 LDS image ([col][Kpad]) once, in workspace. Hot kernels
// stage them with GLOBAL_LOAD_ASYNC_TO_LDS_B128 (ASYNCcnt path) instead of
// per-block cvt+ds_store.
//
// Workspace: 190,976 B of packed f16 weights + bias, then hm + agg
// (2 * 50000 * 128 f32)  ->  ~51.4 MiB total.
// ---------------------------------------------------------------------------

#define N_NODES 50000
#define N_EDGES 800000
#define HID     128
#define EDIM    64

#define KP   136   // 128 + 8 halves pad   (col stride 272B -> bank spread)
#define KP1  72    // 64 + 8 halves pad
#define KPU  264   // 256 + 8 halves pad

// packed f16 weight image offsets (in halves) inside workspace
#define H_NWF   0                       // 128*KP   = 17408
#define H_BT1   17408                   // 128*KP1  =  9216
#define H_BT2   26624                   // 128*KP   = 17408
#define H_BTU1  44032                   // 128*KPU  = 33792
#define H_BTW2  77824                   // 128*KP   = 17408
#define H_TOTAL 95232                   // halves; *2 = 190464 B (16B aligned)

typedef __attribute__((ext_vector_type(16))) _Float16 v16h;
typedef __attribute__((ext_vector_type(8)))  _Float16 v8h;
typedef __attribute__((ext_vector_type(8)))  float    v8f;

#define DEVINL static __device__ __forceinline__

// --- WMMA step:  C += A(16x32 f16) x B(32x16 f16) ---------------------------
DEVINL v8f wmma_step(v16h a, v16h b, v8f c) {
  return __builtin_amdgcn_wmma_f32_16x16x32_f16(
      /*neg_a=*/false, a, /*neg_b=*/false, b,
      /*c_mod=*/(short)0, c, /*reuse_a=*/false, /*reuse_b=*/false);
}

// --- async block copy global(f16 image) -> LDS, 16B per lane per issue ------
// Generic LDS pointer low 32 bits == wave-relative LDS byte address (flat
// aperture mapping), which is what VDST of the async op wants.
DEVINL void async_copy_to_lds(_Float16* lds_dst, const _Float16* __restrict__ src,
                              int halves, int tid) {
  const int chunks = halves >> 3;                 // 8 halves = 16 B
  for (int i = tid; i < chunks; i += 256) {
    unsigned loff = (unsigned)(unsigned long long)(uintptr_t)(lds_dst + i * 8);
    const void* g = (const void*)(src + i * 8);
    asm volatile("global_load_async_to_lds_b128 %0, %1, off"
                 :: "v"(loff), "v"(g)
                 : "memory");
  }
}
DEVINL void async_wait0() {
  asm volatile("s_wait_asynccnt 0x0" ::: "memory");
}

// --- pack 2x8 contiguous f32 -> v16h (A-fragment halves) --------------------
DEVINL v16h pack16(const float* __restrict__ p0, const float* __restrict__ p1) {
  const float4 a0 = ((const float4*)p0)[0];
  const float4 a1 = ((const float4*)p0)[1];
  const float4 b0 = ((const float4*)p1)[0];
  const float4 b1 = ((const float4*)p1)[1];
  v16h r;
  r[0]=(_Float16)a0.x; r[1]=(_Float16)a0.y; r[2]=(_Float16)a0.z; r[3]=(_Float16)a0.w;
  r[4]=(_Float16)a1.x; r[5]=(_Float16)a1.y; r[6]=(_Float16)a1.z; r[7]=(_Float16)a1.w;
  r[8]=(_Float16)b0.x; r[9]=(_Float16)b0.y; r[10]=(_Float16)b0.z; r[11]=(_Float16)b0.w;
  r[12]=(_Float16)b1.x; r[13]=(_Float16)b1.y; r[14]=(_Float16)b1.z; r[15]=(_Float16)b1.w;
  return r;
}

// --- A fragment (16-bit 16x32 ISA layout) from a global f32 row -------------
DEVINL v16h a_global(const float* __restrict__ rowptr, int kt, int lane) {
  const int kb = kt * 32 + ((lane & 16) ? 8 : 0);
  return pack16(rowptr + kb, rowptr + kb + 16);
}

// --- A fragment from LDS f16 tile T[16][kpad] -------------------------------
DEVINL v16h a_lds(const _Float16* __restrict__ T, int kpad, int kt, int lane) {
  const int m  = lane & 15;
  const int kb = kt * 32 + ((lane & 16) ? 8 : 0);
  const _Float16* p = T + m * kpad + kb;
  v8h lo = *(const v8h*)p;
  v8h hi = *(const v8h*)(p + 16);
  return __builtin_shufflevector(lo, hi, 0,1,2,3,4,5,6,7,8,9,10,11,12,13,14,15);
}

// --- B fragment (16-bit 32x16 ISA layout) from LDS Bt[ncols][kpad] ----------
DEVINL v16h b_lds(const _Float16* __restrict__ Bt, int kpad, int nt, int kt, int lane) {
  const int n  = nt * 16 + (lane & 15);
  const int kb = kt * 32 + ((lane & 16) ? 8 : 0);
  const _Float16* p = Bt + n * kpad + kb;
  v8h lo = *(const v8h*)p;
  v8h hi = *(const v8h*)(p + 16);
  return __builtin_shufflevector(lo, hi, 0,1,2,3,4,5,6,7,8,9,10,11,12,13,14,15);
}

// ============================================================================
// Kernel 1: fuse weights + pack ALL B matrices into f16 LDS images.
//   NWfh [n][k] = sum_c node_W[k][c] * msg_W1[c][n]           (K=128)
//   Bt1h [n][k] = sum_c edge_W[k][c] * msg_W1[128+c][n]       (K=64)
//   bias [n]    = msg_b1[n] + node_b.W1h[:,n] + edge_b.W1e[:,n]
//   Bt2h [n][k] = msg_W2[k][n]     BtU1h[n][k] = upd_W1[k][n]   (K=256)
//   BtW2h[n][k] = upd_W2[k][n]
// ============================================================================
__global__ __launch_bounds__(256) void fuse_pack_kernel(
    const float* __restrict__ nW, const float* __restrict__ nb,
    const float* __restrict__ eW, const float* __restrict__ eb,
    const float* __restrict__ mW1, const float* __restrict__ mb1,
    const float* __restrict__ mW2, const float* __restrict__ uW1,
    const float* __restrict__ uW2,
    _Float16* __restrict__ wh, float* __restrict__ bias) {
  const int idx = blockIdx.x * 256 + threadIdx.x;
  if (idx < 16384) {                       // fused node path weight
    const int k = idx >> 7, n = idx & 127;
    float s = 0.f;
    for (int c = 0; c < 128; ++c) s += nW[k * 128 + c] * mW1[c * 128 + n];
    wh[H_NWF + n * KP + k] = (_Float16)s;
  } else if (idx < 24576) {                // fused edge path weight
    const int i2 = idx - 16384;
    const int k = i2 >> 7, n = i2 & 127;
    float s = 0.f;
    for (int c = 0; c < 128; ++c) s += eW[k * 128 + c] * mW1[(128 + c) * 128 + n];
    wh[H_BT1 + n * KP1 + k] = (_Float16)s;
  } else if (idx < 24704) {                // fused bias (f32)
    const int n = idx - 24576;
    float s = mb1[n];
    for (int c = 0; c < 128; ++c)
      s += nb[c] * mW1[c * 128 + n] + eb[c] * mW1[(128 + c) * 128 + n];
    bias[n] = s;
  } else if (idx < 41088) {                // pack msg_W2
    const int i = idx - 24704;
    const int k = i >> 7, n = i & 127;
    wh[H_BT2 + n * KP + k] = (_Float16)mW2[i];
  } else if (idx < 73856) {                // pack upd_W1 (256 x 128)
    const int i = idx - 41088;
    const int k = i >> 7, n = i & 127;
    wh[H_BTU1 + n * KPU + k] = (_Float16)uW1[i];
  } else if (idx < 90240) {                // pack upd_W2
    const int i = idx - 73856;
    const int k = i >> 7, n = i & 127;
    wh[H_BTW2 + n * KP + k] = (_Float16)uW2[i];
  }
}

// ============================================================================
// Kernel 2: hm = node_feats @ NWf + bias    [50000 x 128], K=128
// ============================================================================
__global__ __launch_bounds__(256) void hm_kernel(
    const float* __restrict__ nf, const _Float16* __restrict__ NWfh,
    const float* __restrict__ bias, float* __restrict__ hm) {
  extern __shared__ _Float16 sh[];            // Bt: 128 cols x KP
  const int tid = threadIdx.x;
  async_copy_to_lds(sh, NWfh, 128 * KP, tid);
  async_wait0();
  __syncthreads();

  const int wid = tid >> 5, lane = tid & 31;
  const int m = lane & 15, hi8 = (lane & 16) ? 8 : 0;
  const int tile = blockIdx.x * 8 + wid;      // 3125 tiles (50000/16 exact)
  if (tile >= 3125) return;
  const int rowbase = tile * 16;
  const float* rowptr = nf + (size_t)(rowbase + m) * 128;

  v16h a[4];
#pragma unroll
  for (int kt = 0; kt < 4; ++kt) a[kt] = a_global(rowptr, kt, lane);

#pragma unroll
  for (int nt = 0; nt < 8; ++nt) {
    v8f c = {};
#pragma unroll
    for (int kt = 0; kt < 4; ++kt) c = wmma_step(a[kt], b_lds(sh, KP, nt, kt, lane), c);
    const int n = nt * 16 + m;
    const float bn = bias[n];
#pragma unroll
    for (int i = 0; i < 8; ++i)
      hm[(size_t)(rowbase + i + hi8) * 128 + n] = c[i] + bn;
  }
}

// ============================================================================
// Kernel 3: edge path. Per wave: one 16-edge tile.
//   GEMM1 (K=64) + hm[src] gather + ReLU -> LDS t -> GEMM2 (K=128)
//   -> f32 atomic scatter-add into agg[dst].
// ============================================================================
__global__ __launch_bounds__(256) void edge_kernel(
    const float* __restrict__ ef, const int* __restrict__ eidx,
    const _Float16* __restrict__ Bt1h, const _Float16* __restrict__ Bt2h,
    const float* __restrict__ hm, const float* __restrict__ b2,
    float* __restrict__ agg) {
  extern __shared__ _Float16 sh[];
  _Float16* Bt1  = sh;                         // 128 x KP1
  _Float16* Bt2  = Bt1 + 128 * KP1;            // 128 x KP
  _Float16* Tall = Bt2 + 128 * KP;             // 8 waves x 16 x KP

  const int tid = threadIdx.x;
  async_copy_to_lds(Bt1, Bt1h, 128 * KP1, tid);
  async_copy_to_lds(Bt2, Bt2h, 128 * KP, tid);

  const int wid = tid >> 5, lane = tid & 31;
  const int m = lane & 15, hi8 = (lane & 16) ? 8 : 0;
  const int tile  = blockIdx.x * 8 + wid;      // 50000 tiles exactly
  const int ebase = tile * 16;
  _Float16* T = Tall + wid * 16 * KP;

  // stream the next tile of edge features through the prefetcher (gfx1250)
  {
    const size_t pfrow = (size_t)((ebase + 128 < N_EDGES) ? ebase + 128 : ebase) + m;
    __builtin_prefetch(ef + pfrow * EDIM, 0, 1);
  }

  // overlap: A-tile + indices load while the async weight copy is in flight
  const float* rowptr = ef + (size_t)(ebase + m) * EDIM;
  v16h a[2];
  a[0] = a_global(rowptr, 0, lane);
  a[1] = a_global(rowptr, 1, lane);

  int srcr[8], dstr[8];
#pragma unroll
  for (int i = 0; i < 8; ++i) {
    const int r = i + hi8;
    srcr[i] = eidx[ebase + r];
    dstr[i] = eidx[N_EDGES + ebase + r];
  }

  async_wait0();
  __syncthreads();

  // ---- GEMM1: pre = ef @ W1e  (+ hm[src] gather, + ReLU) -> LDS t ----------
#pragma unroll
  for (int nt = 0; nt < 8; ++nt) {
    v8f c = {};
    c = wmma_step(a[0], b_lds(Bt1, KP1, nt, 0, lane), c);
    c = wmma_step(a[1], b_lds(Bt1, KP1, nt, 1, lane), c);
    const int n = nt * 16 + m;
#pragma unroll
    for (int i = 0; i < 8; ++i) {
      float v = c[i] + hm[(size_t)srcr[i] * 128 + n];   // L2-resident gather
      v = v > 0.f ? v : 0.f;
      T[(i + hi8) * KP + n] = (_Float16)v;
    }
  }

  // ---- GEMM2: m = t @ msg_W2 + b2 -> atomic scatter-add --------------------
  v16h a2[4];
#pragma unroll
  for (int kt = 0; kt < 4; ++kt) a2[kt] = a_lds(T, KP, kt, lane);

#pragma unroll
  for (int nt = 0; nt < 8; ++nt) {
    v8f c = {};
#pragma unroll
    for (int kt = 0; kt < 4; ++kt) c = wmma_step(a2[kt], b_lds(Bt2, KP, nt, kt, lane), c);
    const int n = nt * 16 + m;
    const float bn = b2[n];
#pragma unroll
    for (int i = 0; i < 8; ++i)
      unsafeAtomicAdd(&agg[(size_t)dstr[i] * 128 + n], c[i] + bn);
  }
}

// ============================================================================
// Kernel 4: out = relu([nf, agg] @ upd_W1 + b1) @ upd_W2 + b2    (K=256, 128)
// ============================================================================
__global__ __launch_bounds__(256) void upd_kernel(
    const float* __restrict__ nf, const float* __restrict__ agg,
    const _Float16* __restrict__ BtU1h, const _Float16* __restrict__ BtW2h,
    const float* __restrict__ ub1, const float* __restrict__ ub2,
    float* __restrict__ out) {
  extern __shared__ _Float16 sh[];
  _Float16* BtU1 = sh;                         // 128 x KPU (K = 256)
  _Float16* BtW2 = BtU1 + 128 * KPU;           // 128 x KP
  _Float16* Tall = BtW2 + 128 * KP;            // 8 x 16 x KP

  const int tid = threadIdx.x;
  async_copy_to_lds(BtU1, BtU1h, 128 * KPU, tid);
  async_copy_to_lds(BtW2, BtW2h, 128 * KP, tid);
  async_wait0();
  __syncthreads();

  const int wid = tid >> 5, lane = tid & 31;
  const int m = lane & 15, hi8 = (lane & 16) ? 8 : 0;
  const int tile = blockIdx.x * 8 + wid;
  if (tile >= 3125) return;
  const int rowbase = tile * 16;
  _Float16* T = Tall + wid * 16 * KP;

  const float* nrow = nf  + (size_t)(rowbase + m) * 128;
  const float* arow = agg + (size_t)(rowbase + m) * 128;
  v16h a[8];
#pragma unroll
  for (int kt = 0; kt < 4; ++kt) {
    a[kt]     = a_global(nrow, kt, lane);   // K tiles 0..3 : node_feats
    a[4 + kt] = a_global(arow, kt, lane);   // K tiles 4..7 : agg
  }

#pragma unroll
  for (int nt = 0; nt < 8; ++nt) {
    v8f c = {};
#pragma unroll
    for (int kt = 0; kt < 8; ++kt) c = wmma_step(a[kt], b_lds(BtU1, KPU, nt, kt, lane), c);
    const int n = nt * 16 + m;
    const float bn = ub1[n];
#pragma unroll
    for (int i = 0; i < 8; ++i) {
      float v = c[i] + bn;
      v = v > 0.f ? v : 0.f;
      T[(i + hi8) * KP + n] = (_Float16)v;
    }
  }

  v16h a2[4];
#pragma unroll
  for (int kt = 0; kt < 4; ++kt) a2[kt] = a_lds(T, KP, kt, lane);

#pragma unroll
  for (int nt = 0; nt < 8; ++nt) {
    v8f c = {};
#pragma unroll
    for (int kt = 0; kt < 4; ++kt) c = wmma_step(a2[kt], b_lds(BtW2, KP, nt, kt, lane), c);
    const int n = nt * 16 + m;
    const float bn = ub2[n];
#pragma unroll
    for (int i = 0; i < 8; ++i)
      out[(size_t)(rowbase + i + hi8) * 128 + n] = c[i] + bn;
  }
}

// ============================================================================
extern "C" void kernel_launch(void* const* d_in, const int* in_sizes, int n_in,
                              void* d_out, int out_size, void* d_ws, size_t ws_size,
                              hipStream_t stream) {
  const float* node_feats = (const float*)d_in[0];
  const int*   edge_idx   = (const int*)  d_in[1];
  const float* edge_feats = (const float*)d_in[2];
  const float* node_W  = (const float*)d_in[3];
  const float* node_b  = (const float*)d_in[4];
  const float* edge_W  = (const float*)d_in[5];
  const float* edge_b  = (const float*)d_in[6];
  const float* msg_W1  = (const float*)d_in[7];
  const float* msg_b1  = (const float*)d_in[8];
  const float* msg_W2  = (const float*)d_in[9];
  const float* msg_b2  = (const float*)d_in[10];
  const float* upd_W1  = (const float*)d_in[11];
  const float* upd_b1  = (const float*)d_in[12];
  const float* upd_W2  = (const float*)d_in[13];
  const float* upd_b2  = (const float*)d_in[14];

  _Float16* wh  = (_Float16*)d_ws;                          // packed f16 images
  float* bias = (float*)((char*)d_ws + (size_t)H_TOTAL * 2);          // 128 f32
  float* hm   = (float*)((char*)d_ws + (size_t)H_TOTAL * 2 + 512);    // N*H f32
  float* agg  = hm + (size_t)N_NODES * HID;

  // agg is accumulated atomically -> must start at zero every call.
  hipMemsetAsync(agg, 0, (size_t)N_NODES * HID * sizeof(float), stream);

  fuse_pack_kernel<<<353, 256, 0, stream>>>(node_W, node_b, edge_W, edge_b,
                                            msg_W1, msg_b1, msg_W2, upd_W1,
                                            upd_W2, wh, bias);

  hm_kernel<<<391, 256, 128 * KP * 2, stream>>>(node_feats, wh + H_NWF, bias, hm);

  const size_t lds_edge = (size_t)(128 * KP1 + 128 * KP + 8 * 16 * KP) * 2;
  edge_kernel<<<N_EDGES / 128, 256, lds_edge, stream>>>(
      edge_feats, edge_idx, wh + H_BT1, wh + H_BT2, hm, msg_b2, agg);

  const size_t lds_upd = (size_t)(128 * KPU + 128 * KP + 8 * 16 * KP) * 2;
  upd_kernel<<<391, 256, lds_upd, stream>>>(node_feats, agg, wh + H_BTU1,
                                            wh + H_BTW2, upd_b1, upd_b2,
                                            (float*)d_out);
}
